// APFA_51539607796
// MI455X (gfx1250) — matrix operand order repaired and checked by reading, not compiled, not run on hardware
//
#include <hip/hip_runtime.h>
#include <hip/hip_bf16.h>
#include <math.h>

typedef __attribute__((ext_vector_type(16))) _Float16 v16h;
typedef __attribute__((ext_vector_type(8)))  _Float16 v8h;
typedef __attribute__((ext_vector_type(2)))  _Float16 h2;
typedef __attribute__((ext_vector_type(8)))  float    v8f;

// Problem constants (from reference): B=4, H=W=64, N=4096, C=256, h=8, d=32,
// SR=4 -> PL=256, WS=3 -> LOCAL=9.

// ---------------------------------------------------------------------------
// WMMA GEMM: C[M,N] = A[M,K] @ B[K,N] (+bias) (+optional exact GELU).
// 256 threads = 8 waves in a 4(m) x 2(n) grid; each wave owns an
// (MR*16) x (NR*16) micro-tile -> MR*NR v_wmma_f32_16x16x32_f16 per K-step.
// Block tile: BM = 64*MR, BN = 32*NR; K step 32.
// Big GEMMs use MR=2,NR=4 (128x128 tile, 8 WMMA/wave/K-step).
// BTRANS=true: B given row-major as [N,K] (computes A @ B^T).
// Batched over blockIdx.z with per-tensor batch strides.
// f32 -> f16 conversion happens exactly once per element during LDS staging
// (packed half2 ds_store_b32); TDM/async staging would force the convert on
// the consumer side, so the VGPR-roundtrip path is the right one here.
// ---------------------------------------------------------------------------
template<int EPI, bool BTRANS, bool HASB, int MR, int NR>
__global__ __launch_bounds__(256)
void gemm_wmma(const float* __restrict__ A, int lda, long long bsA,
               const float* __restrict__ Bm, int ldb, long long bsB,
               const float* __restrict__ bias,
               float* __restrict__ Cm, int ldc, long long bsC,
               int K)
{
    constexpr int BM = 64 * MR;
    constexpr int BN = 32 * NR;
    constexpr int APAIR = BM * 16 / 256;   // half2 pairs per thread for A tile
    constexpr int BPAIR = BN * 16 / 256;   // half2 pairs per thread for B tile

    __shared__ _Float16 sA[BM * 32];       // [m][k], row stride 32 halfs
    __shared__ _Float16 sB[BN * 32];       // transposed: [n][k], row stride 32

    const int tid  = threadIdx.x;
    const int lane = tid & 31;
    const int wave = tid >> 5;
    const int wm   = wave >> 1;            // 0..3
    const int wn   = wave & 1;             // 0..1
    const int m0 = blockIdx.y * BM;
    const int n0 = blockIdx.x * BN;
    const float* Ab = A  + (long long)blockIdx.z * bsA;
    const float* Bb = Bm + (long long)blockIdx.z * bsB;
    float*       Cb = Cm + (long long)blockIdx.z * bsC;

    // 16-bit A 16x32 fragment: lane L -> row M=L&15; lanes 0-15 K 0..15,
    // lanes 16-31 K 16..31 (contiguous, matching B).
    const int kbA = (lane >= 16) ? 16 : 0;
    // 16-bit B 32x16 fragment: lane L -> col N=L&15; lanes 0-15 K 0..15,
    // lanes 16-31 K 16..31 (contiguous in the transposed sB row).
    const int kbB = (lane >= 16) ? 16 : 0;

    v8f acc[MR][NR] = {};

    int k0 = 0;
    do {
        // Prefetch next A K-tile, one 128B row per thread, WGP scope
        // (locality 3 -> pull into all cache levels; lowers to global_prefetch_b8)
        if (k0 + 32 < K && tid < BM)
            __builtin_prefetch(&Ab[(m0 + tid) * lda + k0 + 32], 0, 3);

        // Stage A tile: float2 load -> packed half2 -> one ds_store_b32
        #pragma unroll
        for (int i = 0; i < APAIR; ++i) {
            int idx = (i << 8) + tid;
            int k2 = idx & 15, r = idx >> 4;
            const float2 f = *(const float2*)&Ab[(m0 + r) * lda + k0 + 2 * k2];
            h2 hp; hp.x = (_Float16)f.x; hp.y = (_Float16)f.y;
            *(h2*)&sA[r * 32 + 2 * k2] = hp;
        }
        // Stage B tile transposed into sB[n][k]
        #pragma unroll
        for (int i = 0; i < BPAIR; ++i) {
            int idx = (i << 8) + tid;
            if (BTRANS) {
                int k2 = idx & 15, nb = idx >> 4;
                const float2 f = *(const float2*)&Bb[(n0 + nb) * ldb + k0 + 2 * k2];
                h2 hp; hp.x = (_Float16)f.x; hp.y = (_Float16)f.y;
                *(h2*)&sB[nb * 32 + 2 * k2] = hp;
            } else {
                int nb = idx % BN, k2 = idx / BN;
                float b0 = Bb[(k0 + 2 * k2)     * ldb + n0 + nb];
                float b1 = Bb[(k0 + 2 * k2 + 1) * ldb + n0 + nb];
                h2 hp; hp.x = (_Float16)b0; hp.y = (_Float16)b1;
                *(h2*)&sB[nb * 32 + 2 * k2] = hp;
            }
        }
        __syncthreads();

        v16h af[MR];
        v16h bf[NR];
        #pragma unroll
        for (int mi = 0; mi < MR; ++mi) {
            int mrow = ((wm * MR + mi) << 4) + (lane & 15);
            af[mi] = *(const v16h*)&sA[mrow * 32 + kbA];
        }
        #pragma unroll
        for (int ni = 0; ni < NR; ++ni) {
            int ncol = ((wn * NR + ni) << 4) + (lane & 15);
            bf[ni] = *(const v16h*)&sB[ncol * 32 + kbB];
        }
        #pragma unroll
        for (int mi = 0; mi < MR; ++mi)
            #pragma unroll
            for (int ni = 0; ni < NR; ++ni)
                acc[mi][ni] = __builtin_amdgcn_wmma_f32_16x16x32_f16(
                    false, af[mi], false, bf[ni], (short)0, acc[mi][ni],
                    false, false);
        __syncthreads();
        k0 += 32;
    } while (k0 < K);

    // Epilogue: f32 C/D layout -> lane L col N=L&15; VGPR g row M=g+(L>=16?8:0)
    #pragma unroll
    for (int mi = 0; mi < MR; ++mi) {
        const int rb = m0 + ((wm * MR + mi) << 4) + ((lane >= 16) ? 8 : 0);
        #pragma unroll
        for (int ni = 0; ni < NR; ++ni) {
            const int colb = n0 + ((wn * NR + ni) << 4) + (lane & 15);
            float bv = 0.0f;
            if (HASB) bv = bias[colb];
            float* cp = Cb + rb * ldc + colb;
            #pragma unroll
            for (int g = 0; g < 8; ++g) {
                float v = acc[mi][ni][g] + bv;
                if (EPI == 1)   // exact GELU
                    v = 0.5f * v * (1.0f + erff(v * 0.70710678118654752f));
                cp[g * ldc] = v;
            }
        }
    }
}

// ---------------------------------------------------------------------------
// 4x4 average pool over the 64x64 grid + LayerNorm over C=256.
// One block per pooled pixel (1024 blocks), 256 threads = one channel each.
// ---------------------------------------------------------------------------
__global__ __launch_bounds__(256)
void pool_ln_kernel(const float* __restrict__ xsg,
                    const float* __restrict__ g, const float* __restrict__ b,
                    float* __restrict__ out)
{
    int blk = blockIdx.x;            // bb*256 + ph*16 + pw
    int bb = blk >> 8;
    int ph = (blk >> 4) & 15;
    int pw = blk & 15;
    int c = threadIdx.x;

    float s = 0.f;
    #pragma unroll
    for (int i = 0; i < 4; ++i)
        #pragma unroll
        for (int j = 0; j < 4; ++j)
            s += xsg[((bb * 4096) + (ph * 4 + i) * 64 + (pw * 4 + j)) * 256 + c];
    float v = s * 0.0625f;

    __shared__ float red[256];
    red[c] = v; __syncthreads();
    for (int o = 128; o > 0; o >>= 1) { if (c < o) red[c] += red[c + o]; __syncthreads(); }
    float mu = red[0] * (1.f / 256.f);
    __syncthreads();
    float dv = v - mu;
    red[c] = dv * dv; __syncthreads();
    for (int o = 128; o > 0; o >>= 1) { if (c < o) red[c] += red[c + o]; __syncthreads(); }
    float var = red[0] * (1.f / 256.f);
    out[blk * 256 + c] = dv * rsqrtf(var + 1e-5f) * g[c] + b[c];
}

// ---------------------------------------------------------------------------
// cpb MLP: t[l,hh] = relu(rct[l]@W1 + b1) @ W2 + b2, stored transposed as
// biasT[hh*1024 + l] for cheap gather in the softmax kernel. Tiny (17 MFLOP).
// ---------------------------------------------------------------------------
__global__ __launch_bounds__(256)
void cpb_kernel(const float* __restrict__ rct,
                const float* __restrict__ w1, const float* __restrict__ b1,
                const float* __restrict__ w2, const float* __restrict__ b2,
                float* __restrict__ biasT)
{
    int idx = blockIdx.x * 256 + threadIdx.x;   // 8192 = 1024 * 8
    int l = idx >> 3, hh = idx & 7;
    float c0 = rct[l * 2], c1 = rct[l * 2 + 1];
    float acc = 0.f;
    for (int j = 0; j < 512; ++j) {
        float hd = fmaxf(c0 * w1[j] + c1 * w1[512 + j] + b1[j], 0.f);
        acc += hd * w2[j * 8 + hh];
    }
    biasT[hh * 1024 + l] = acc + b2[hh];
}

// ---------------------------------------------------------------------------
// q prep: per (b,h,n): L2-normalize over d=32, then
// q_scaled = (q_norm + query_embedding) * softplus(temperature[h]) * seq_scale(n)
// seq_scale(n) = log(count3x3(n) + PL). Output layout (B,h,N,d).
// ---------------------------------------------------------------------------
__global__ __launch_bounds__(256)
void prep_q_kernel(const float* __restrict__ qlin, const float* __restrict__ qe,
                   const float* __restrict__ temp,
                   float* __restrict__ qn, float* __restrict__ qs)
{
    int idx = blockIdx.x * 256 + threadIdx.x;   // B*h*N = 131072
    int n  = idx & 4095;
    int hh = (idx >> 12) & 7;
    int bb = idx >> 15;
    const float* src = qlin + (bb * 4096 + n) * 256 + hh * 32;
    float v[32]; float ss = 0.f;
    #pragma unroll
    for (int d = 0; d < 32; ++d) { v[d] = src[d]; ss += v[d] * v[d]; }
    float inv = rsqrtf(ss + 1e-12f);
    int r = n >> 6, c = n & 63;
    int cnt = ((r > 0) + 1 + (r < 63)) * ((c > 0) + 1 + (c < 63));
    float scale = log1pf(expf(temp[hh])) * logf((float)cnt + 256.0f);
    int o = idx * 32;
    #pragma unroll
    for (int d = 0; d < 32; ++d) {
        float nv = v[d] * inv;
        qn[o + d] = nv;
        qs[o + d] = (nv + qe[hh * 32 + d]) * scale;
    }
}

// k_local L2-norm: kv layout (B,N,2h*d); heads 0..7 = K. Output (B,h,N,d).
__global__ __launch_bounds__(256)
void prep_k_kernel(const float* __restrict__ kv, float* __restrict__ kn)
{
    int idx = blockIdx.x * 256 + threadIdx.x;   // 131072
    int n  = idx & 4095;
    int hh = (idx >> 12) & 7;
    int bb = idx >> 15;
    const float* src = kv + (bb * 4096 + n) * 512 + hh * 32;
    float v[32]; float ss = 0.f;
    #pragma unroll
    for (int d = 0; d < 32; ++d) { v[d] = src[d]; ss += v[d] * v[d]; }
    float inv = rsqrtf(ss + 1e-12f);
    int o = idx * 32;
    #pragma unroll
    for (int d = 0; d < 32; ++d) kn[o + d] = v[d] * inv;
}

// pooled KV prep: kp = l2norm(k_pool), v_pool copied contiguous (B,h,PL,d).
__global__ __launch_bounds__(256)
void prep_kp_kernel(const float* __restrict__ kvp,
                    float* __restrict__ kp, float* __restrict__ vpool)
{
    int idx = blockIdx.x * 256 + threadIdx.x;    // B*h*PL = 8192
    int pp = idx & 255;
    int hh = (idx >> 8) & 7;
    int bb = idx >> 11;
    const float* srck = kvp + (bb * 256 + pp) * 512 + hh * 32;
    const float* srcv = srck + 256;              // (8+hh)*32 = hh*32 + 256
    float v[32]; float ss = 0.f;
    #pragma unroll
    for (int d = 0; d < 32; ++d) { v[d] = srck[d]; ss += v[d] * v[d]; }
    float inv = rsqrtf(ss + 1e-12f);
    int o = idx * 32;
    #pragma unroll
    for (int d = 0; d < 32; ++d) { kp[o + d] = v[d] * inv; vpool[o + d] = srcv[d]; }
}

// 3x3 local attention logits (+rpb, OOB mask). aloc row stride 16.
__global__ __launch_bounds__(256)
void attn_local_kernel(const float* __restrict__ qs, const float* __restrict__ kn,
                       const float* __restrict__ rpb, float* __restrict__ aloc)
{
    int idx = blockIdx.x * 256 + threadIdx.x;   // 131072
    int n  = idx & 4095;
    int hh = (idx >> 12) & 7;
    int bb = idx >> 15;
    int r = n >> 6, c = n & 63;
    float qv[32];
    const float* q = qs + idx * 32;
    #pragma unroll
    for (int d = 0; d < 32; ++d) qv[d] = q[d];
    const float* knb = kn + (bb * 8 + hh) * 4096 * 32;
    #pragma unroll
    for (int l = 0; l < 9; ++l) {
        int rr = r + l / 3 - 1, cc = c + l % 3 - 1;
        float val = -1e9f;
        if (rr >= 0 && rr < 64 && cc >= 0 && cc < 64) {
            const float* kk = knb + (rr * 64 + cc) * 32;
            float s = 0.f;
            #pragma unroll
            for (int d = 0; d < 32; ++d) s += qv[d] * kk[d];
            val = s + rpb[hh * 9 + l];
        }
        aloc[(idx << 4) + l] = val;
    }
}

// Fused: gather pool bias, softmax over concat(9 local, 256 pool).
// One wave32 per row; 8 pool values per lane; local on lanes 0..8.
__global__ __launch_bounds__(256)
void softmax_kernel(float* __restrict__ aloc, float* __restrict__ apool,
                    const float* __restrict__ biasT, const int* __restrict__ rpi)
{
    int row = blockIdx.x * 8 + (threadIdx.x >> 5);  // 131072 rows
    int lane = threadIdx.x & 31;
    int n  = row & 4095;
    int hh = (row >> 12) & 7;
    float* pl = apool + (long long)row * 256;
    const int* rp = rpi + n * 256;
    float pv[8];
    float mx = -1e30f;
    #pragma unroll
    for (int i = 0; i < 8; ++i) {
        int p = i * 32 + lane;
        float v = pl[p] + biasT[hh * 1024 + rp[p]];
        pv[i] = v; mx = fmaxf(mx, v);
    }
    float lcl = (lane < 9) ? aloc[(row << 4) + lane] : -1e30f;
    mx = fmaxf(mx, lcl);
    #pragma unroll
    for (int o = 16; o > 0; o >>= 1) mx = fmaxf(mx, __shfl_xor(mx, o, 32));
    float sum = 0.f;
    #pragma unroll
    for (int i = 0; i < 8; ++i) { pv[i] = expf(pv[i] - mx); sum += pv[i]; }
    float le = (lane < 9) ? expf(lcl - mx) : 0.f;
    sum += le;
    #pragma unroll
    for (int o = 16; o > 0; o >>= 1) sum += __shfl_xor(sum, o, 32);
    float inv = 1.f / sum;
    #pragma unroll
    for (int i = 0; i < 8; ++i) pl[i * 32 + lane] = pv[i] * inv;
    if (lane < 9) aloc[(row << 4) + lane] = le * inv;
}

// attn_local += q_norm @ learnable_tokens + learnable_bias; aggregate local V;
// add x_pool; write combined (B,N,C) for the final projection GEMM.
__global__ __launch_bounds__(256)
void xlocal_kernel(const float* __restrict__ qn, const float* __restrict__ aloc,
                   const float* __restrict__ ltok, const float* __restrict__ lb,
                   const float* __restrict__ kv, const float* __restrict__ xpool,
                   float* __restrict__ comb)
{
    int idx = blockIdx.x * 256 + threadIdx.x;   // 131072
    int n  = idx & 4095;
    int hh = (idx >> 12) & 7;
    int bb = idx >> 15;
    int r = n >> 6, c = n & 63;
    float qv[32];
    const float* q = qn + idx * 32;
    #pragma unroll
    for (int d = 0; d < 32; ++d) qv[d] = q[d];
    float a9[9];
    #pragma unroll
    for (int l = 0; l < 9; ++l) {
        const float* tk = ltok + hh * 32 * 9 + l;   // (h,d,9)
        float s = 0.f;
        #pragma unroll
        for (int d = 0; d < 32; ++d) s += qv[d] * tk[d * 9];
        a9[l] = aloc[(idx << 4) + l] + s + lb[hh * 9 + l];
    }
    float ov[32];
    const float* xp = xpool + idx * 32;
    #pragma unroll
    for (int d = 0; d < 32; ++d) ov[d] = xp[d];
    #pragma unroll
    for (int l = 0; l < 9; ++l) {
        int rr = r + l / 3 - 1, cc = c + l % 3 - 1;
        if (rr < 0 || rr > 63 || cc < 0 || cc > 63) continue;   // v_uf is 0 there
        const float* vp = kv + (bb * 4096 + rr * 64 + cc) * 512 + 256 + hh * 32;
        float a = a9[l];
        #pragma unroll
        for (int d = 0; d < 32; ++d) ov[d] += a * vp[d];
    }
    float* dst = comb + (bb * 4096 + n) * 256 + hh * 32;
    #pragma unroll
    for (int d = 0; d < 32; ++d) dst[d] = ov[d];
}

// ---------------------------------------------------------------------------
extern "C" void kernel_launch(void* const* d_in, const int* in_sizes, int n_in,
                              void* d_out, int out_size, void* d_ws, size_t ws_size,
                              hipStream_t stream)
{
    (void)in_sizes; (void)n_in; (void)out_size; (void)ws_size;
    const float* x    = (const float*)d_in[0];
    const float* rct  = (const float*)d_in[1];
    const int*   rpi  = (const int*)  d_in[2];
    const float* q_w  = (const float*)d_in[3];
    const float* q_b  = (const float*)d_in[4];
    const float* kv_w = (const float*)d_in[5];
    const float* kv_b = (const float*)d_in[6];
    const float* temp = (const float*)d_in[7];
    const float* qe   = (const float*)d_in[8];
    const float* sr_w = (const float*)d_in[9];
    const float* sr_b = (const float*)d_in[10];
    const float* ng   = (const float*)d_in[11];
    const float* nbias= (const float*)d_in[12];
    const float* c1w  = (const float*)d_in[13];
    const float* c1b  = (const float*)d_in[14];
    const float* c2w  = (const float*)d_in[15];
    const float* c2b  = (const float*)d_in[16];
    const float* rpb  = (const float*)d_in[17];
    const float* ltok = (const float*)d_in[18];
    const float* lbias= (const float*)d_in[19];
    const float* pw   = (const float*)d_in[20];
    const float* pb   = (const float*)d_in[21];
    float* out = (float*)d_out;

    // Workspace partition (floats)
    float* p = (float*)d_ws;
    float* q_lin = p; p += (size_t)16384 * 256;
    float* kv    = p; p += (size_t)16384 * 512;
    float* xsg   = p; p += (size_t)16384 * 256;
    float* xs_ln = p; p += (size_t)1024 * 256;
    float* kvp   = p; p += (size_t)1024 * 512;
    float* q_nrm = p; p += (size_t)4 * 8 * 4096 * 32;
    float* q_sc  = p; p += (size_t)4 * 8 * 4096 * 32;
    float* kn    = p; p += (size_t)4 * 8 * 4096 * 32;
    float* kp    = p; p += (size_t)4 * 8 * 256 * 32;
    float* vpool = p; p += (size_t)4 * 8 * 256 * 32;
    float* biasT = p; p += (size_t)8 * 1024;
    float* aloc  = p; p += (size_t)4 * 8 * 4096 * 16;
    float* apool = p; p += (size_t)4 * 8 * 4096 * 256;
    float* xpool = p; p += (size_t)4 * 8 * 4096 * 32;
    float* comb  = p; p += (size_t)16384 * 256;

    dim3 blk(256);

    // q = x@q_w + q_b                         (16384 x 256, K=256)
    gemm_wmma<0,false,true,2,4><<<dim3(2, 128, 1), blk, 0, stream>>>(
        x, 256, 0, q_w, 256, 0, q_b, q_lin, 256, 0, 256);
    // kv = x@kv_w + kv_b                      (16384 x 512, K=256)
    gemm_wmma<0,false,true,2,4><<<dim3(4, 128, 1), blk, 0, stream>>>(
        x, 256, 0, kv_w, 512, 0, kv_b, kv, 512, 0, 256);
    // xsg = gelu(x@sr_w + sr_b)               (16384 x 256, K=256)
    gemm_wmma<1,false,true,2,4><<<dim3(2, 128, 1), blk, 0, stream>>>(
        x, 256, 0, sr_w, 256, 0, sr_b, xsg, 256, 0, 256);
    // 4x4 avg pool + LayerNorm -> xs_ln (1024 x 256)
    pool_ln_kernel<<<dim3(1024), blk, 0, stream>>>(xsg, ng, nbias, xs_ln);
    // kvp = xs_ln@kv_w + kv_b                 (1024 x 512, K=256)
    gemm_wmma<0,false,true,2,4><<<dim3(4, 8, 1), blk, 0, stream>>>(
        xs_ln, 256, 0, kv_w, 512, 0, kv_b, kvp, 512, 0, 256);
    // cpb MLP -> biasT[h][1024]
    cpb_kernel<<<dim3(32), blk, 0, stream>>>(rct, c1w, c1b, c2w, c2b, biasT);
    // q / k / pooled-kv prep
    prep_q_kernel<<<dim3(512), blk, 0, stream>>>(q_lin, qe, temp, q_nrm, q_sc);
    prep_k_kernel<<<dim3(512), blk, 0, stream>>>(kv, kn);
    prep_kp_kernel<<<dim3(32), blk, 0, stream>>>(kvp, kp, vpool);
    // local attention logits
    attn_local_kernel<<<dim3(512), blk, 0, stream>>>(q_sc, kn, rpb, aloc);
    // attn_pool = q_scaled @ kp^T  (batched over b,h=32: 4096x256, K=32)
    gemm_wmma<0,true,false,2,4><<<dim3(2, 32, 32), blk, 0, stream>>>(
        q_sc, 32, (long long)4096 * 32,
        kp,   32, (long long)256 * 32, nullptr,
        apool, 256, (long long)4096 * 256, 32);
    // pool-bias gather + softmax over 9+256
    softmax_kernel<<<dim3(16384), blk, 0, stream>>>(aloc, apool, biasT, rpi);
    // x_pool = attn_pool @ v_pool  (batched: 4096x32, K=256)
    gemm_wmma<0,false,false,2,1><<<dim3(1, 32, 32), blk, 0, stream>>>(
        apool, 256, (long long)4096 * 256,
        vpool, 32,  (long long)256 * 32, nullptr,
        xpool, 32,  (long long)4096 * 32, 256);
    // local aggregation + combine -> (B,N,C)
    xlocal_kernel<<<dim3(512), blk, 0, stream>>>(
        q_nrm, aloc, ltok, lbias, kv, xpool, comb);
    // out = comb@proj_w + proj_b              (16384 x 256, K=256)
    gemm_wmma<0,false,true,2,4><<<dim3(2, 128, 1), blk, 0, stream>>>(
        comb, 256, 0, pw, 256, 0, pb, out, 256, 0, 256);
}
